// GptInferenceModel_10118942949733
// MI455X (gfx1250) — matrix operand order repaired and checked by reading, not compiled, not run on hardware
//
#include <hip/hip_runtime.h>
#include <hip/hip_bf16.h>
#include <stdint.h>

typedef __attribute__((ext_vector_type(16))) _Float16 v16h;
typedef __attribute__((ext_vector_type(8)))  float    v8f;

#define VSZ   32000
#define DM    768
#define NH    12
#define DHD   64
#define NLAY  6
#define BB    8
#define LL0   128
#define TT    384
#define KTOP  10
#define VSID  5
#define FFD   3072

// ----------------------------- device helpers -----------------------------

__device__ __forceinline__ unsigned short f2h_bits(float x) {
  union { _Float16 h; unsigned short u; } c;
  c.h = (_Float16)x;
  return c.u;
}

__device__ __forceinline__ float pe_val(int pos, int d) {
  float e = (float)(2 * (d >> 1)) / (float)DM;
  float denom = powf(10000.f, e);
  float ang = (float)pos / denom;
  return (d & 1) ? cosf(ang) : sinf(ang);
}

__device__ __forceinline__ unsigned long long splitmix64(unsigned long long x) {
  x += 0x9E3779B97F4A7C15ull;
  x = (x ^ (x >> 30)) * 0xBF58476D1CE4E5B9ull;
  x = (x ^ (x >> 27)) * 0x94D049BB133111EBull;
  return x ^ (x >> 31);
}

template<int NT>
__device__ __forceinline__ float blk_sum(float v, float* red) {
  int t = threadIdx.x;
  red[t] = v; __syncthreads();
  #pragma unroll
  for (int s = NT >> 1; s > 0; s >>= 1) {
    if (t < s) red[t] += red[t + s];
    __syncthreads();
  }
  float r = red[0]; __syncthreads();
  return r;
}

template<int NT>
__device__ __forceinline__ float blk_max(float v, float* red) {
  int t = threadIdx.x;
  red[t] = v; __syncthreads();
  #pragma unroll
  for (int s = NT >> 1; s > 0; s >>= 1) {
    if (t < s) red[t] = fmaxf(red[t], red[t + s]);
    __syncthreads();
  }
  float r = red[0]; __syncthreads();
  return r;
}

// ----------------------------- cast fp32 -> f16 -----------------------------

__global__ void k_cast_f16(const float* __restrict__ s, _Float16* __restrict__ d, long n) {
  long i = (long)blockIdx.x * blockDim.x + threadIdx.x;
  if (i < n) d[i] = (_Float16)s[i];
}

// ----------------------------- WMMA GEMM -----------------------------------
// C[m,n] = sum_k A[m,k] * B[k,n]   (A fp32 in, B f16 in, C fp32 out)
// A row addressing:  arow = (m / a_rpb) * a_bs + (m % a_rpb) * lda
// C row addressing:  crow = (m / c_rpb) * c_bs + (m % c_rpb) * ldc
// bt == 0 : B is [K,N] row-major f16 (weights); synchronous transposing stage
// bt == 1 : B is [N,K] row-major f16 (embedding, logits = x @ emb^T);
//           staged with GLOBAL_LOAD_ASYNC_TO_LDS_B128 (ASYNCcnt path)
// Each wave computes a 16x64 strip: one A fragment feeds 4 WMMAs (reuse_a).
// Requires: N % 256 == 0, K % 32 == 0.
__global__ void __launch_bounds__(128, 2)
k_gemm(const float* __restrict__ A, int a_rpb, long a_bs, int lda,
       const _Float16* __restrict__ Bw, int ldb, int bt,
       float* __restrict__ C, int c_rpb, long c_bs, int ldc,
       int M, int N, int K, int relu)
{
  __shared__ __align__(16) unsigned short As[16 * 32];           // 1 KB (shared by 4 waves)
  __shared__ __align__(16) unsigned short Bs[4][4][16 * 32];     // 16 KB: [wave][ntile][n][k]

  const int tid  = threadIdx.x;
  const int wave = tid >> 5;
  const int lane = tid & 31;
  const int m0 = blockIdx.y * 16;
  const int n0 = blockIdx.x * 256 + wave * 64;                   // 64-wide strip per wave

  v8f acc[4];
  #pragma unroll
  for (int j = 0; j < 4; ++j) acc[j] = (v8f){0.f,0.f,0.f,0.f,0.f,0.f,0.f,0.f};

  for (int k0 = 0; k0 < K; k0 += 32) {
    // ---- stage B strip (32 x 64) for this wave ----
    if (bt == 0) {
      int kk = lane;                         // k within tile
      const _Float16* src = Bw + (long)(k0 + kk) * ldb + n0;     // 64 contiguous f16
      if (k0 + kk + 32 < K) __builtin_prefetch((const void*)(src + (long)32 * ldb), 0, 1);
      #pragma unroll
      for (int nt = 0; nt < 4; ++nt) {
        unsigned short tmp[16];
        *(uint4*)&tmp[0] = *(const uint4*)(src + nt * 16);
        *(uint4*)&tmp[8] = *(const uint4*)(src + nt * 16 + 8);
        #pragma unroll
        for (int n = 0; n < 16; ++n) Bs[wave][nt][n * 32 + kk] = tmp[n];
      }
    } else {
      // async copy: lane pair (n, half) pulls 32 contiguous bytes of emb row
      int n  = lane >> 1;
      int kh = (lane & 1) * 16;
      #pragma unroll
      for (int nt = 0; nt < 4; ++nt) {
        int gn = n0 + nt * 16 + n;
        if (gn < N) {
          const _Float16* src = Bw + (long)gn * ldb + k0 + kh;
          unsigned lds_addr = (unsigned)(size_t)&Bs[wave][nt][n * 32 + kh];
          unsigned long long gaddr = (unsigned long long)(size_t)src;
          asm volatile(
            "global_load_async_to_lds_b128 %0, %1, off\n"
            "global_load_async_to_lds_b128 %0, %1, off offset:16\n"
            :: "v"(lds_addr), "v"(gaddr) : "memory");
        }
      }
    }
    // ---- stage A tile (16 x 32) fp32 -> f16, 4 floats per thread ----
    {
      int e = tid * 4;
      int r = e >> 5, c = e & 31;
      int gm = m0 + r;
      float4 f = make_float4(0.f, 0.f, 0.f, 0.f);
      if (gm < M) {
        long arow = (long)(gm / a_rpb) * a_bs + (long)(gm % a_rpb) * lda;
        f = *(const float4*)(A + arow + k0 + c);
      }
      unsigned short* dst = &As[r * 32 + c];
      dst[0] = f2h_bits(f.x); dst[1] = f2h_bits(f.y);
      dst[2] = f2h_bits(f.z); dst[3] = f2h_bits(f.w);
    }
    __syncthreads();
    if (bt == 1) asm volatile("s_wait_asynccnt 0x0" ::: "memory");

    // ---- gather wave32 WMMA fragments from LDS ----
    union { unsigned u[8]; v16h v; } fa;
    {
      int m   = lane & 15;
      int klo = (lane >> 4) * 8;     // A: lanes 0-15 hold K 0-7/16-23; lanes 16-31 hold 8-15/24-31
      #pragma unroll
      for (int j = 0; j < 4; ++j) {
        fa.u[j]     = *(const unsigned*)&As[m * 32 + klo + 2 * j];
        fa.u[4 + j] = *(const unsigned*)&As[m * 32 + 16 + klo + 2 * j];
      }
    }
    {
      int n  = lane & 15;
      int kq = (lane >> 4) * 8;      // B: lanes 0-15 hold K 0-7/16-23; lanes 16-31 hold 8-15/24-31
      union { unsigned u[8]; v16h v; } fb[4];
      #pragma unroll
      for (int nt = 0; nt < 4; ++nt) {
        #pragma unroll
        for (int j = 0; j < 4; ++j) {
          fb[nt].u[j]     = *(const unsigned*)&Bs[wave][nt][n * 32 + kq + 2 * j];
          fb[nt].u[4 + j] = *(const unsigned*)&Bs[wave][nt][n * 32 + 16 + kq + 2 * j];
        }
      }
      // 4 WMMAs share one A fragment; hint A-operand reuse on all but the last
      acc[0] = __builtin_amdgcn_wmma_f32_16x16x32_f16(false, fa.v, false, fb[0].v,
                                                      (short)0, acc[0], true,  false);
      acc[1] = __builtin_amdgcn_wmma_f32_16x16x32_f16(false, fa.v, false, fb[1].v,
                                                      (short)0, acc[1], true,  false);
      acc[2] = __builtin_amdgcn_wmma_f32_16x16x32_f16(false, fa.v, false, fb[2].v,
                                                      (short)0, acc[2], true,  false);
      acc[3] = __builtin_amdgcn_wmma_f32_16x16x32_f16(false, fa.v, false, fb[3].v,
                                                      (short)0, acc[3], false, false);
    }
    __syncthreads();
  }

  // ---- store 16x64 strip: lane = N col, vgpr r = M row (+8 for hi lanes) ----
  int mb = (lane >> 4) * 8;
  #pragma unroll
  for (int nt = 0; nt < 4; ++nt) {
    int n = n0 + nt * 16 + (lane & 15);
    if (n < N) {
      #pragma unroll
      for (int r = 0; r < 8; ++r) {
        int gm = m0 + mb + r;
        if (gm < M) {
          float v = acc[nt][r];
          if (relu) v = v > 0.f ? v : 0.f;
          long crow = (long)(gm / c_rpb) * c_bs + (long)(gm % c_rpb) * ldc;
          C[crow + n] = v;
        }
      }
    }
  }
}

// ----------------------------- embedding ------------------------------------

__global__ void __launch_bounds__(256)
k_embed_pre(const int* __restrict__ tokens, const float* __restrict__ emb,
            float* __restrict__ xemb, float* __restrict__ embc,
            int* __restrict__ tok, int* __restrict__ ord)
{
  int bl = blockIdx.x;                 // b*LL0 + l
  int b = bl / LL0, l = bl % LL0;
  int t = tokens[bl];
  for (int d = threadIdx.x; d < DM; d += 256) {
    float v = emb[(long)t * DM + d] + pe_val(0, d);
    xemb[(long)bl * DM + d] = v;
    embc[((long)b * TT + l) * DM + d] = v;
  }
  if (threadIdx.x == 0) {
    tok[b * TT + l] = t;
    ord[b * TT + l] = 0;
  }
}

__global__ void __launch_bounds__(256)
k_embed_dec(const int* __restrict__ tok, const int* __restrict__ ord,
            const float* __restrict__ emb, float* __restrict__ xc,
            float* __restrict__ embc, int p)
{
  int b = blockIdx.x;
  int t = tok[b * TT + p];
  int o = ord[b * TT + p];
  for (int d = threadIdx.x; d < DM; d += 256) {
    float v = emb[(long)t * DM + d] + pe_val(o, d);
    xc[(long)b * DM + d] = v;
    embc[((long)b * TT + p) * DM + d] = v;
  }
}

// ----------------------------- attention ------------------------------------

__global__ void __launch_bounds__(64)
k_attn_pre(const float* __restrict__ Q, const float* __restrict__ Kc,
           const float* __restrict__ Vc, float* __restrict__ O)
{
  int idx = blockIdx.x;                // b*NH*LL0 + h*LL0 + l
  int l = idx % LL0;
  int h = (idx / LL0) % NH;
  int b = idx / (LL0 * NH);
  __shared__ float sc[LL0];
  __shared__ float red[64];
  const float* q = Q + ((long)b * LL0 + l) * DM + h * DHD;
  int n = l + 1;
  int t = threadIdx.x;
  float mx = -3.4e38f;
  for (int k = t; k < n; k += 64) {
    const float* kr = Kc + ((long)b * TT + k) * DM + h * DHD;
    float s = 0.f;
    #pragma unroll 16
    for (int d = 0; d < DHD; ++d) s += q[d] * kr[d];
    s *= 0.125f;                       // 1/sqrt(64)
    sc[k] = s;
    mx = fmaxf(mx, s);
  }
  mx = blk_max<64>(mx, red);
  float sum = 0.f;
  for (int k = t; k < n; k += 64) { float e = __expf(sc[k] - mx); sc[k] = e; sum += e; }
  sum = blk_sum<64>(sum, red);
  float inv = 1.f / sum;
  __syncthreads();
  if (t < DHD) {
    float o = 0.f;
    for (int k = 0; k < n; ++k) o += sc[k] * Vc[((long)b * TT + k) * DM + h * DHD + t];
    O[((long)b * LL0 + l) * DM + h * DHD + t] = o * inv;
  }
}

__global__ void __launch_bounds__(128)
k_attn_dec(const float* __restrict__ Q, const float* __restrict__ Kc,
           const float* __restrict__ Vc, float* __restrict__ O, int p)
{
  int h = blockIdx.x % NH;
  int b = blockIdx.x / NH;
  __shared__ float sc[TT];
  __shared__ float red[128];
  const float* q = Q + (long)b * DM + h * DHD;
  int n = p + 1;
  int t = threadIdx.x;
  float mx = -3.4e38f;
  for (int k = t; k < n; k += 128) {
    const float* kr = Kc + ((long)b * TT + k) * DM + h * DHD;
    float s = 0.f;
    #pragma unroll 16
    for (int d = 0; d < DHD; ++d) s += q[d] * kr[d];
    s *= 0.125f;
    sc[k] = s;
    mx = fmaxf(mx, s);
  }
  mx = blk_max<128>(mx, red);
  float sum = 0.f;
  for (int k = t; k < n; k += 128) { float e = __expf(sc[k] - mx); sc[k] = e; sum += e; }
  sum = blk_sum<128>(sum, red);
  float inv = 1.f / sum;
  __syncthreads();
  if (t < DHD) {
    float o = 0.f;
    for (int k = 0; k < n; ++k) o += sc[k] * Vc[((long)b * TT + k) * DM + h * DHD + t];
    O[(long)b * DM + h * DHD + t] = o * inv;
  }
}

// ----------------------------- layernorm ------------------------------------
// y = LN(X + A) * g + bb ; written to Y (compact, may alias X) and optionally
// Y2 with row addressing (row/rpb)*bs + (row%rpb)*DM.
__global__ void __launch_bounds__(256)
k_ln(const float* __restrict__ X, const float* __restrict__ A,
     const float* __restrict__ g, const float* __restrict__ bb,
     float* __restrict__ Y, float* __restrict__ Y2, int rpb, long bs)
{
  __shared__ float buf[DM];
  __shared__ float red[256];
  int row = blockIdx.x;
  long ro = (long)row * DM;
  int t = threadIdx.x;
  float s = 0.f;
  for (int d = t; d < DM; d += 256) {
    float v = X[ro + d] + A[ro + d];
    buf[d] = v;
    s += v;
  }
  s = blk_sum<256>(s, red);
  float mu = s / (float)DM;
  float s2 = 0.f;
  for (int d = t; d < DM; d += 256) { float v = buf[d] - mu; s2 += v * v; }
  s2 = blk_sum<256>(s2, red);
  float inv = rsqrtf(s2 / (float)DM + 1e-6f);
  long y2o = (long)(row / rpb) * bs + (long)(row % rpb) * DM;
  for (int d = t; d < DM; d += 256) {
    float y = (buf[d] - mu) * inv * g[d] + bb[d];
    if (Y)  Y[ro + d] = y;
    if (Y2) Y2[y2o + d] = y;
  }
}

// ----------------------------- vocab membership mask -------------------------

__global__ void k_zero_u32(unsigned* __restrict__ m, int n) {
  int i = blockIdx.x * blockDim.x + threadIdx.x;
  if (i < n) m[i] = 0u;
}

__global__ void k_set_mask(const int* __restrict__ ids, int n, unsigned* __restrict__ m) {
  int i = blockIdx.x * blockDim.x + threadIdx.x;
  if (i < n) {
    int t = ids[i];
    if (t >= 0 && t < VSZ) atomicOr(&m[t >> 5], 1u << (t & 31));
  }
}

// ----------------------------- top-k gumbel sampling -------------------------

__global__ void __launch_bounds__(256)
k_sample(const float* __restrict__ logits, const unsigned* __restrict__ amask,
         int* __restrict__ tok, int* __restrict__ ord, int p,
         unsigned long long seed)
{
  int b = blockIdx.x;
  const float* lg = logits + (long)b * VSZ;
  float lv[KTOP]; int li[KTOP];
  #pragma unroll
  for (int j = 0; j < KTOP; ++j) { lv[j] = -3.4e38f; li[j] = -1; }
  for (int i = threadIdx.x; i < VSZ; i += 256) {
    float v = lg[i];
    if (v > lv[KTOP - 1]) {
      int j = KTOP - 1;
      while (j > 0 && lv[j - 1] < v) { lv[j] = lv[j - 1]; li[j] = li[j - 1]; --j; }
      lv[j] = v; li[j] = i;
    }
  }
  __shared__ float sv[256 * KTOP];
  __shared__ int   si[256 * KTOP];
  #pragma unroll
  for (int j = 0; j < KTOP; ++j) {
    sv[threadIdx.x * KTOP + j] = lv[j];
    si[threadIdx.x * KTOP + j] = li[j];
  }
  __syncthreads();
  if (threadIdx.x == 0) {
    float gv[KTOP]; int gi[KTOP];
    #pragma unroll
    for (int j = 0; j < KTOP; ++j) { gv[j] = -3.4e38f; gi[j] = -1; }
    for (int e = 0; e < 256 * KTOP; ++e) {
      float v = sv[e]; int id = si[e];
      if (id >= 0 && v > gv[KTOP - 1]) {
        int j = KTOP - 1;
        while (j > 0 && gv[j - 1] < v) { gv[j] = gv[j - 1]; gi[j] = gi[j - 1]; --j; }
        gv[j] = v; gi[j] = id;
      }
    }
    int best = 0; float bestv = -3.4e38f;
    #pragma unroll
    for (int j = 0; j < KTOP; ++j) {
      unsigned long long h = splitmix64(seed ^ ((unsigned long long)(b + 1) * 0x9E3779B97F4A7C15ull)
                                            ^ ((unsigned long long)(j + 1) * 0xD1B54A32D192ED03ull));
      float u = (float)(((h >> 11) + 1.0) * (1.0 / 9007199254740994.0));
      u = fminf(fmaxf(u, 1e-7f), 1.f - 1e-7f);
      float gg = -__logf(-__logf(u));
      float c = gv[j] + gg;
      if (c > bestv) { bestv = c; best = j; }
    }
    int nt = gi[best] < 0 ? 0 : gi[best];
    int tp = tok[b * TT + p];
    if ((amask[tp >> 5] >> (tp & 31)) & 1u) nt = VSID;
    tok[b * TT + p + 1] = nt;
    ord[b * TT + p + 1] = ord[b * TT + p] + ((nt == VSID) ? 1 : 0);
  }
}

// ----------------------------- host orchestration ----------------------------

extern "C" void kernel_launch(void* const* d_in, const int* in_sizes, int n_in,
                              void* d_out, int out_size, void* d_ws, size_t ws_size,
                              hipStream_t stream)
{
  (void)n_in; (void)out_size; (void)ws_size;
  const int*   tokens  = (const int*)d_in[0];
  const int*   att_ids = (const int*)d_in[1];
  const float* emb     = (const float*)d_in[2];
  const float* Wq      = (const float*)d_in[3];
  const float* Wk      = (const float*)d_in[4];
  const float* Wv      = (const float*)d_in[5];
  const float* Wo      = (const float*)d_in[6];
  const float* W1      = (const float*)d_in[7];
  const float* W2      = (const float*)d_in[8];
  const float* g1      = (const float*)d_in[9];
  const float* b1      = (const float*)d_in[10];
  const float* g2      = (const float*)d_in[11];
  const float* b2      = (const float*)d_in[12];
  const int n_att = in_sizes[1];

  int* tok = (int*)d_out;                          // (B, T) int32 output buffer

  // ---- workspace bump allocator (deterministic layout) ----
  char* base = (char*)d_ws;
  size_t off = 0;
  auto alloc = [&](size_t bytes) -> char* {
    char* p = base + off;
    off = (off + bytes + 255) & ~(size_t)255;
    return p;
  };
  const size_t WSZ  = (size_t)NLAY * DM * DM;
  const size_t FSZ  = (size_t)NLAY * DM * FFD;
  const size_t BTD  = (size_t)BB * TT * DM;
  const size_t LBTD = (size_t)NLAY * BTD;
  const int    MPRE = BB * LL0;

  _Float16* wq16  = (_Float16*)alloc(sizeof(_Float16) * WSZ);
  _Float16* wk16  = (_Float16*)alloc(sizeof(_Float16) * WSZ);
  _Float16* wv16  = (_Float16*)alloc(sizeof(_Float16) * WSZ);
  _Float16* wo16  = (_Float16*)alloc(sizeof(_Float16) * WSZ);
  _Float16* w116  = (_Float16*)alloc(sizeof(_Float16) * FSZ);
  _Float16* w216  = (_Float16*)alloc(sizeof(_Float16) * FSZ);
  _Float16* emb16 = (_Float16*)alloc(sizeof(_Float16) * (size_t)VSZ * DM);
  float* embc   = (float*)alloc(sizeof(float) * BTD);
  float* layc   = (float*)alloc(sizeof(float) * LBTD);
  float* Kc     = (float*)alloc(sizeof(float) * LBTD);
  float* Vc     = (float*)alloc(sizeof(float) * LBTD);
  float* xemb   = (float*)alloc(sizeof(float) * (size_t)MPRE * DM);
  float* qb     = (float*)alloc(sizeof(float) * (size_t)MPRE * DM);
  float* ab     = (float*)alloc(sizeof(float) * (size_t)MPRE * DM);
  float* ob     = (float*)alloc(sizeof(float) * (size_t)MPRE * DM);
  float* x1b    = (float*)alloc(sizeof(float) * (size_t)MPRE * DM);
  float* ff2b   = (float*)alloc(sizeof(float) * (size_t)MPRE * DM);
  float* ff1b   = (float*)alloc(sizeof(float) * (size_t)MPRE * FFD);
  float* xc     = (float*)alloc(sizeof(float) * (size_t)BB * DM);
  float* logits = (float*)alloc(sizeof(float) * (size_t)BB * VSZ);
  int*   ordb   = (int*)alloc(sizeof(int) * (size_t)BB * TT);
  unsigned* amask = (unsigned*)alloc(sizeof(unsigned) * 1024);

  auto cast = [&](const float* s, _Float16* d, long n) {
    k_cast_f16<<<dim3((unsigned)((n + 255) / 256)), dim3(256), 0, stream>>>(s, d, n);
  };
  auto gemm = [&](const float* A, int a_rpb, long a_bs, int lda,
                  const _Float16* Bw, int ldb, int bt,
                  float* C, int c_rpb, long c_bs, int ldc,
                  int M, int N, int K, int relu) {
    dim3 grid((unsigned)(N / 256), (unsigned)((M + 15) / 16));
    k_gemm<<<grid, 128, 0, stream>>>(A, a_rpb, a_bs, lda, Bw, ldb, bt,
                                     C, c_rpb, c_bs, ldc, M, N, K, relu);
  };

  // ---- one-time weight / embedding casts to f16 (L2-resident ~134 MB) ----
  cast(Wq, wq16, (long)WSZ);
  cast(Wk, wk16, (long)WSZ);
  cast(Wv, wv16, (long)WSZ);
  cast(Wo, wo16, (long)WSZ);
  cast(W1, w116, (long)FSZ);
  cast(W2, w216, (long)FSZ);
  cast(emb, emb16, (long)VSZ * DM);

  // ---- vocab membership bitmask ----
  k_zero_u32<<<4, 256, 0, stream>>>(amask, 1000);
  k_set_mask<<<(n_att + 255) / 256, 256, 0, stream>>>(att_ids, n_att, amask);

  // ---- prefill ----
  k_embed_pre<<<MPRE, 256, 0, stream>>>(tokens, emb, xemb, embc, tok, ordb);
  for (int i = 0; i < NLAY; ++i) {
    const _Float16* wq_i = wq16 + (size_t)i * DM * DM;
    const _Float16* wk_i = wk16 + (size_t)i * DM * DM;
    const _Float16* wv_i = wv16 + (size_t)i * DM * DM;
    const _Float16* wo_i = wo16 + (size_t)i * DM * DM;
    const _Float16* w1_i = w116 + (size_t)i * DM * FFD;
    const _Float16* w2_i = w216 + (size_t)i * FFD * DM;
    float* Kci = Kc + (size_t)i * BTD;
    float* Vci = Vc + (size_t)i * BTD;
    float* Lci = layc + (size_t)i * BTD;
    // kv == q input == xemb (reference: kv = x_emb for layer 0, else previous layer out)
    gemm(xemb, MPRE, 0, DM, wq_i, DM, 0, qb,  MPRE, 0, DM, MPRE, DM, DM, 0);
    gemm(xemb, MPRE, 0, DM, wk_i, DM, 0, Kci, LL0, (long)TT * DM, DM, MPRE, DM, DM, 0);
    gemm(xemb, MPRE, 0, DM, wv_i, DM, 0, Vci, LL0, (long)TT * DM, DM, MPRE, DM, DM, 0);
    k_attn_pre<<<BB * NH * LL0, 64, 0, stream>>>(qb, Kci, Vci, ab);
    gemm(ab, MPRE, 0, DM, wo_i, DM, 0, ob, MPRE, 0, DM, MPRE, DM, DM, 0);
    k_ln<<<MPRE, 256, 0, stream>>>(xemb, ob, g1 + i * DM, b1 + i * DM, x1b, nullptr, 1, 0);
    gemm(x1b, MPRE, 0, DM, w1_i, FFD, 0, ff1b, MPRE, 0, FFD, MPRE, FFD, DM, 1);
    gemm(ff1b, MPRE, 0, FFD, w2_i, DM, 0, ff2b, MPRE, 0, DM, MPRE, DM, FFD, 0);
    k_ln<<<MPRE, 256, 0, stream>>>(x1b, ff2b, g2 + i * DM, b2 + i * DM, xemb,
                                   Lci, LL0, (long)TT * DM);
  }
  // first sample from last prompt position
  gemm(xemb + (size_t)(LL0 - 1) * DM, 1, (long)LL0 * DM, DM, emb16, DM, 1,
       logits, BB, 0, VSZ, BB, VSZ, DM, 0);
  k_sample<<<BB, 256, 0, stream>>>(logits, amask, tok, ordb, LL0 - 1,
                                   0x5DEECE66Dull * 1ull + 42ull);

  // ---- decode: 255 serial steps ----
  for (int s = 0; s < TT - LL0 - 1; ++s) {
    int p = LL0 + s;
    k_embed_dec<<<BB, 256, 0, stream>>>(tok, ordb, emb, xc, embc, p);
    for (int i = 0; i < NLAY; ++i) {
      const _Float16* wq_i = wq16 + (size_t)i * DM * DM;
      const _Float16* wk_i = wk16 + (size_t)i * DM * DM;
      const _Float16* wv_i = wv16 + (size_t)i * DM * DM;
      const _Float16* wo_i = wo16 + (size_t)i * DM * DM;
      const _Float16* w1_i = w116 + (size_t)i * DM * FFD;
      const _Float16* w2_i = w216 + (size_t)i * FFD * DM;
      float* Kci = Kc + (size_t)i * BTD;
      float* Vci = Vc + (size_t)i * BTD;
      float* Lci = layc + (size_t)i * BTD;
      // append new K/V row at position p (cached rows are mathematically invariant)
      gemm(xc, BB, 0, DM, wk_i, DM, 0, Kci + (size_t)p * DM, 1, (long)TT * DM, DM, BB, DM, DM, 0);
      gemm(xc, BB, 0, DM, wv_i, DM, 0, Vci + (size_t)p * DM, 1, (long)TT * DM, DM, BB, DM, DM, 0);
      gemm(xc, BB, 0, DM, wq_i, DM, 0, qb, BB, 0, DM, BB, DM, DM, 0);
      k_attn_dec<<<BB * NH, 128, 0, stream>>>(qb, Kci, Vci, ab, p);
      gemm(ab, BB, 0, DM, wo_i, DM, 0, ob, BB, 0, DM, BB, DM, DM, 0);
      k_ln<<<BB, 256, 0, stream>>>(xc, ob, g1 + i * DM, b1 + i * DM, x1b, nullptr, 1, 0);
      gemm(x1b, BB, 0, DM, w1_i, FFD, 0, ff1b, BB, 0, FFD, BB, FFD, DM, 1);
      gemm(ff1b, BB, 0, FFD, w2_i, DM, 0, ff2b, BB, 0, DM, BB, DM, FFD, 0);
      k_ln<<<BB, 256, 0, stream>>>(x1b, ff2b, g2 + i * DM, b2 + i * DM, xc,
                                   Lci + (size_t)p * DM, 1, (long)TT * DM);
    }
    gemm(xc, BB, 0, DM, emb16, DM, 1, logits, BB, 0, VSZ, BB, VSZ, DM, 0);
    k_sample<<<BB, 256, 0, stream>>>(logits, amask, tok, ordb, p,
                                     0x5DEECE66Dull * (unsigned long long)(s + 2) + 42ull);
  }
}
